// GraphSAGE_62569083568743
// MI455X (gfx1250) — matrix operand order repaired and checked by reading, not compiled, not run on hardware
//
#include <hip/hip_runtime.h>
#include <hip/hip_bf16.h>
#include <math.h>

// ---------------------------------------------------------------------------
// GraphSAGE (3-layer, mean aggregator) for MI455X / gfx1250.
//   - edge scatter-add stays L2-resident (h + msum = 102 MB < 192 MB L2)
//   - fused [h|hmean] @ [Ws;Wn] GEMM (K=512) via v_wmma_f32_16x16x32_bf16
//   - each wave: 2 m-tiles x 4 n-tiles -> B fragments amortized over 2 WMMAs
// ---------------------------------------------------------------------------

typedef __bf16 bf16_t;
typedef __attribute__((ext_vector_type(16))) __bf16 v16bf;
typedef __attribute__((ext_vector_type(8)))  float  v8f;

#define N_NODES 50000
#define N_EDGES 800000
#define FEAT    256
#define KBIG    512      // fused K = 2*FEAT
#define MTILES  3125     // 50000 / 16 exactly
#define NCLS    47

// ---------------- utility kernels ----------------

__global__ void zero_f32(float* __restrict__ p, int n4) {
    int i = blockIdx.x * blockDim.x + threadIdx.x;
    if (i < n4) reinterpret_cast<float4*>(p)[i] = make_float4(0.f, 0.f, 0.f, 0.f);
}

__global__ void deg_accum(const int* __restrict__ dst, float* __restrict__ deg) {
    int e = blockIdx.x * blockDim.x + threadIdx.x;
    if (e < N_EDGES) atomicAdd(deg + dst[e], 1.0f);
}

__global__ void deg_recip(float* __restrict__ deg) {
    int i = blockIdx.x * blockDim.x + threadIdx.x;
    if (i < N_NODES) deg[i] = 1.0f / fmaxf(deg[i], 1.0f);
}

// one float4 of one edge per thread: gather h[src], scatter-add to msum[dst]
__global__ void aggregate(const float* __restrict__ h, const int* __restrict__ src,
                          const int* __restrict__ dst, float* __restrict__ msum) {
    int t = blockIdx.x * blockDim.x + threadIdx.x;
    int e = t >> 6;                 // 64 threads per edge (256 feats / 4)
    if (e >= N_EDGES) return;
    int f = (t & 63) << 2;
    int s = src[e], d = dst[e];
    float4 v = *reinterpret_cast<const float4*>(h + (size_t)s * FEAT + f);
    float* o = msum + (size_t)d * FEAT + f;
    atomicAdd(o + 0, v.x);
    atomicAdd(o + 1, v.y);
    atomicAdd(o + 2, v.z);
    atomicAdd(o + 3, v.w);
}

// Abig[i, 0:256] = bf16(h[i]); Abig[i, 256:512] = bf16(msum[i] * deginv[i])
__global__ void pack_A(const float* __restrict__ h, const float* __restrict__ msum,
                       const float* __restrict__ deginv, bf16_t* __restrict__ A) {
    int idx = blockIdx.x * blockDim.x + threadIdx.x;
    if (idx >= N_NODES * FEAT) return;
    int i = idx >> 8, f = idx & 255;
    A[(size_t)i * KBIG + f]        = (bf16_t)h[idx];
    A[(size_t)i * KBIG + FEAT + f] = (bf16_t)(msum[idx] * deginv[i]);
}

// Pre-swizzle Wbig = [Ws; Wn] (fp32, row-major [512 x dout]) into B-fragment
// order (lane = col N, lanes 0-15 hold K=k0..k0+7,k0+16..k0+23, lanes 16-31
// hold K=k0+8..k0+15,k0+24..k0+31, matching A). Zero-pad n >= dout.
__global__ void pack_W(const float* __restrict__ Ws, const float* __restrict__ Wn,
                       bf16_t* __restrict__ Wp, int NTpad, int dout) {
    int idx   = blockIdx.x * blockDim.x + threadIdx.x;
    int total = 16 * NTpad * 512;           // KT(16) * NTpad frags * 512 elems
    if (idx >= total) return;
    int e    = idx & 15;
    int lane = (idx >> 4) & 31;
    int frag = idx >> 9;
    int nt   = frag % NTpad;
    int kt   = frag / NTpad;
    int k = kt * 32 + ((lane >> 4) << 3) + (e & 7) + ((e >> 3) << 4);    // +8 for upper half-wave
    int n = nt * 16 + (lane & 15);
    float v = 0.0f;
    if (n < dout) v = (k < FEAT) ? Ws[k * dout + n] : Wn[(k - FEAT) * dout + n];
    Wp[idx] = (bf16_t)v;
}

__device__ __forceinline__ float selu_f(float x) {
    const float scale = 1.0507009873554805f;
    const float alpha = 1.6732632423543772f;
    return x > 0.0f ? scale * x : scale * alpha * (__expf(x) - 1.0f);
}

__device__ __forceinline__ void store_tile(float* __restrict__ out,
                                           const float* __restrict__ bias,
                                           int ldo, int dout, int act,
                                           int rBase, int colBase,
                                           v8f a0, v8f a1, v8f a2, v8f a3) {
    v8f accs[4] = {a0, a1, a2, a3};
#pragma unroll
    for (int j = 0; j < 4; ++j) {
        int col = colBase + j * 16;
        if (col < dout) {
            float bb = bias[col];
#pragma unroll
            for (int r = 0; r < 8; ++r) {
                float v = accs[j][r] + bb;
                if (act == 0) v = selu_f(v);
                out[(size_t)(rBase + r) * ldo + col] = v;
            }
        }
    }
}

// ---------------- WMMA GEMM: out[M x dout] = Abig[M x 512] @ Wbig + b ------
// Each wave: 2 m-tiles (32 rows) x 4 n-tiles (64 cols) = 8 v8f accumulators.
// B fragments loaded once per K-step and fed to two WMMAs (both m-tiles).
// 8 waves / block; nGroups = NTpad/4 waves span N; wave-uniform guards only
// (EXEC stays all-1s around every WMMA, as the ISA requires).
__global__ void sage_gemm(const bf16_t* __restrict__ A, const bf16_t* __restrict__ Wp,
                          const float* __restrict__ bias, float* __restrict__ out,
                          int NTpad, int dout, int ldo, int act) {
    int lane    = threadIdx.x & 31;
    int waveId  = threadIdx.x >> 5;
    int nGroups = NTpad >> 2;                    // 4 (dout=256) or 1 (dout=47)
    int waveN   = waveId % nGroups;
    int waveM   = waveId / nGroups;
    int mWaves  = 8 / nGroups;                   // waves along M per block
    int mTile0  = (blockIdx.x * mWaves + waveM) * 2;
    if (mTile0 >= MTILES) return;                // wave-uniform
    int  mTile1 = mTile0 + 1;
    bool live1  = mTile1 < MTILES;
    int  mT1c   = live1 ? mTile1 : (MTILES - 1); // clamp for compute, mask store
    int  ntBase = waveN * 4;

    // A-fragment base: row = mTile*16 + lane%16, K base offset 0 / +8 per half
    const bf16_t* aBase0 =
        A + (size_t)(mTile0 * 16 + (lane & 15)) * KBIG + ((lane >> 4) << 3);
    const bf16_t* aBase1 =
        A + (size_t)(mT1c * 16 + (lane & 15)) * KBIG + ((lane >> 4) << 3);
    const bf16_t* wBase =
        Wp + ((size_t)ntBase * 32 + lane) * 16;  // + kt*NTpad*512 each K-step

    v8f p0 = {}, p1 = {}, p2 = {}, p3 = {};      // m-tile 0 accumulators
    v8f q0 = {}, q1 = {}, q2 = {}, q3 = {};      // m-tile 1 accumulators

    for (int kt = 0; kt < 16; ++kt) {
        union { uint4 u[2]; v16bf v; } af0, af1;
        const uint4* pa0 = reinterpret_cast<const uint4*>(aBase0 + kt * 32);
        const uint4* pa1 = reinterpret_cast<const uint4*>(aBase1 + kt * 32);
        af0.u[0] = pa0[0]; af0.u[1] = pa0[2];    // elems 0-7: K+0..7, 8-15: K+16..23
        af1.u[0] = pa1[0]; af1.u[1] = pa1[2];

        const bf16_t* wk = wBase + (size_t)kt * NTpad * 512;
        {
            union { uint4 u[2]; v16bf v; } b;
            const uint4* pb = reinterpret_cast<const uint4*>(wk);
            b.u[0] = pb[0]; b.u[1] = pb[1];
            p0 = __builtin_amdgcn_wmma_f32_16x16x32_bf16(false, af0.v, false, b.v,
                                                         (short)0, p0, false, false);
            q0 = __builtin_amdgcn_wmma_f32_16x16x32_bf16(false, af1.v, false, b.v,
                                                         (short)0, q0, false, false);
        }
        {
            union { uint4 u[2]; v16bf v; } b;
            const uint4* pb = reinterpret_cast<const uint4*>(wk + 512);
            b.u[0] = pb[0]; b.u[1] = pb[1];
            p1 = __builtin_amdgcn_wmma_f32_16x16x32_bf16(false, af0.v, false, b.v,
                                                         (short)0, p1, false, false);
            q1 = __builtin_amdgcn_wmma_f32_16x16x32_bf16(false, af1.v, false, b.v,
                                                         (short)0, q1, false, false);
        }
        {
            union { uint4 u[2]; v16bf v; } b;
            const uint4* pb = reinterpret_cast<const uint4*>(wk + 1024);
            b.u[0] = pb[0]; b.u[1] = pb[1];
            p2 = __builtin_amdgcn_wmma_f32_16x16x32_bf16(false, af0.v, false, b.v,
                                                         (short)0, p2, false, false);
            q2 = __builtin_amdgcn_wmma_f32_16x16x32_bf16(false, af1.v, false, b.v,
                                                         (short)0, q2, false, false);
        }
        {
            union { uint4 u[2]; v16bf v; } b;
            const uint4* pb = reinterpret_cast<const uint4*>(wk + 1536);
            b.u[0] = pb[0]; b.u[1] = pb[1];
            p3 = __builtin_amdgcn_wmma_f32_16x16x32_bf16(false, af0.v, false, b.v,
                                                         (short)0, p3, false, false);
            q3 = __builtin_amdgcn_wmma_f32_16x16x32_bf16(false, af1.v, false, b.v,
                                                         (short)0, q3, false, false);
        }
    }

    // C/D layout: lane<16 -> rows r+0..7, lane>=16 -> rows r+8..15; col = lane%16
    int rOff    = (lane >> 4) << 3;
    int colBase = ntBase * 16 + (lane & 15);
    store_tile(out, bias, ldo, dout, act, mTile0 * 16 + rOff, colBase, p0, p1, p2, p3);
    if (live1)                                    // wave-uniform store predicate
        store_tile(out, bias, ldo, dout, act, mTile1 * 16 + rOff, colBase, q0, q1, q2, q3);
}

__global__ void softmax47(float* __restrict__ out) {
    int i = blockIdx.x * blockDim.x + threadIdx.x;
    if (i >= N_NODES) return;
    float* row = out + (size_t)i * NCLS;
    float m = -INFINITY;
    for (int j = 0; j < NCLS; ++j) m = fmaxf(m, row[j]);
    float s = 0.0f;
    for (int j = 0; j < NCLS; ++j) s += __expf(row[j] - m);
    float inv = 1.0f / s;
    for (int j = 0; j < NCLS; ++j) row[j] = __expf(row[j] - m) * inv;
}

// ---------------------------------------------------------------------------

extern "C" void kernel_launch(void* const* d_in, const int* in_sizes, int n_in,
                              void* d_out, int out_size, void* d_ws, size_t ws_size,
                              hipStream_t stream) {
    (void)in_sizes; (void)n_in; (void)out_size; (void)ws_size;

    const float* x   = (const float*)d_in[0];
    const int*   src = (const int*)d_in[1];
    const int*   dst = (const int*)d_in[2];
    const float* Wself[3]  = {(const float*)d_in[3], (const float*)d_in[6], (const float*)d_in[9]};
    const float* Wneigh[3] = {(const float*)d_in[4], (const float*)d_in[7], (const float*)d_in[10]};
    const float* bvec[3]   = {(const float*)d_in[5], (const float*)d_in[8], (const float*)d_in[11]};

    // workspace carve (bytes): msum 51.2M | hbuf 51.2M | Abig 51.2M | Wpack 256K | deginv 200K
    char*   w      = (char*)d_ws;
    float*  msum   = (float*)(w);
    float*  hbuf   = (float*)(w + 51200000);
    bf16_t* Abig   = (bf16_t*)(w + 102400000);
    bf16_t* Wpack  = (bf16_t*)(w + 153600000);
    float*  deginv = (float*)(w + 153600000 + 262144);

    // in-degree reciprocal (shared by all layers)
    zero_f32 <<<(N_NODES / 4 + 255) / 256, 256, 0, stream>>>(deginv, N_NODES / 4);
    deg_accum<<<(N_EDGES + 255) / 256, 256, 0, stream>>>(dst, deginv);
    deg_recip<<<(N_NODES + 255) / 256, 256, 0, stream>>>(deginv);

    const int douts[3]  = {256, 256, NCLS};
    const int NTpads[3] = {16, 16, 4};       // n-tiles (padded to multiple of 4)

    const float* hin = x;
    for (int L = 0; L < 3; ++L) {
        int dout = douts[L], NTpad = NTpads[L];

        zero_f32 <<<(N_NODES * FEAT / 4 + 255) / 256, 256, 0, stream>>>(msum, N_NODES * FEAT / 4);
        aggregate<<<(N_EDGES * 64 + 255) / 256, 256, 0, stream>>>(hin, src, dst, msum);
        pack_A   <<<(N_NODES * FEAT + 255) / 256, 256, 0, stream>>>(hin, msum, deginv, Abig);

        int wElems = 16 * NTpad * 512;
        pack_W<<<(wElems + 255) / 256, 256, 0, stream>>>(Wself[L], Wneigh[L], Wpack, NTpad, dout);

        int nGroups = NTpad / 4;
        int mTilesPerBlk = (8 / nGroups) * 2;    // 2 m-tiles per wave
        int blocks  = (MTILES + mTilesPerBlk - 1) / mTilesPerBlk;
        float* outp = (L == 2) ? (float*)d_out : hbuf;
        int    ldo  = (L == 2) ? NCLS : 256;
        int    act  = (L == 2) ? 1 : 0;          // 0 = SELU, 1 = raw logits
        sage_gemm<<<blocks, 256, 0, stream>>>(Abig, Wpack, bvec[L], outp, NTpad, dout, ldo, act);

        hin = hbuf;
    }

    softmax47<<<(N_NODES + 255) / 256, 256, 0, stream>>>((float*)d_out);
}
